// CostVolumePrompt_7842610282726
// MI455X (gfx1250) — matrix operand order repaired and checked by reading, not compiled, not run on hardware
//
#include <hip/hip_runtime.h>
#include <hip/hip_bf16.h>

// CDNA5 (gfx1250) fused cost-volume + softmax + depth/conf kernel.
// One workgroup per (b,h); bf16 WMMA GEMM out of LDS, softmax fused in regs.
// v2.1: two M-stripes per wave (each B fragment feeds 2 WMMAs), exp2-domain
//       softmax via __builtin_amdgcn_exp2f (direct v_exp_f32).

typedef __attribute__((ext_vector_type(16))) __bf16 bf16x16;
typedef __attribute__((ext_vector_type(8)))  float  f32x8;

#define H_ 128
#define W_ 512
#define C_ 128
#define LDS_ROW 272                 // 128 bf16 (256B) + 16B pad -> conflict-free b128 ops
#define A_BASE 0
#define B_BASE (W_ * LDS_ROW)       // 139264
#define SMEM_BYTES (2 * W_ * LDS_ROW)  // 278528 < 320KB WGP LDS

union V16 { uint4 q[2]; bf16x16 v; };

static __device__ inline float fast_exp2(float x) {
    return __builtin_amdgcn_exp2f(x);   // v_exp_f32
}

static __device__ inline unsigned pack_bf16(float a, float b) {
    __bf16 lo = (__bf16)a;
    __bf16 hi = (__bf16)b;
    unsigned short ul = __builtin_bit_cast(unsigned short, lo);
    unsigned short uh = __builtin_bit_cast(unsigned short, hi);
    return (unsigned)ul | ((unsigned)uh << 16);
}

__global__ __launch_bounds__(256, 1)
void costvol_fused_kernel(const float* __restrict__ feat,
                          const float* __restrict__ extri,
                          const float* __restrict__ intri,
                          const float* __restrict__ farp,
                          float* __restrict__ out)
{
    extern __shared__ char smem[];

    const int bh   = blockIdx.x;       // 0..511
    const int b    = bh >> 7;          // 0..3
    const int h    = bh & (H_ - 1);    // 0..127
    const int tid  = threadIdx.x;
    const int lane = tid & 31;
    const int wave = tid >> 5;
    const int half = lane >> 4;        // 0: lanes 0-15, 1: lanes 16-31

    // ---- per-batch scalars: fx * |t_l - t_r| / far ----
    const float fx = intri[(b * 2 + 0) * 9 + 0];
    const float dx = extri[((b * 2 + 0) * 4 + 0) * 4 + 3] - extri[((b * 2 + 1) * 4 + 0) * 4 + 3];
    const float dy = extri[((b * 2 + 0) * 4 + 1) * 4 + 3] - extri[((b * 2 + 1) * 4 + 1) * 4 + 3];
    const float dz = extri[((b * 2 + 0) * 4 + 2) * 4 + 3] - extri[((b * 2 + 1) * 4 + 2) * 4 + 3];
    const float fxb_over_far = fx * sqrtf(dx * dx + dy * dy + dz * dz) / farp[b * 2 + 0];

    // ---- stage L and R slices (C_ x W_) into LDS as bf16, transposed to [w][c] ----
    // item: one (matrix, c-pair, w-pair) -> 2 global float2 loads, 2 packed b32 LDS stores
    // items = 2 * 64 * 256 = 32768 ; / 256 threads = 128 iterations
    for (int it = 0; it < 128; ++it) {
        int item = it * 256 + tid;         // 0..32767
        int v    = item >> 14;             // 0 = left, 1 = right
        int rem  = item & 16383;
        int cp   = rem >> 8;               // 0..63 (c pair)
        int w2   = (rem & 255) * 2;        // 0..510 step 2
        int c    = cp * 2;
        const float* src = feat + ((((size_t)(b * 2 + v)) * C_ + c) * H_ + h) * W_ + w2;
        float2 r0 = *reinterpret_cast<const float2*>(src);                       // row c
        float2 r1 = *reinterpret_cast<const float2*>(src + (size_t)H_ * W_);     // row c+1
        char* dst = smem + (v ? B_BASE : A_BASE) + (size_t)w2 * LDS_ROW + c * 2;
        *reinterpret_cast<unsigned*>(dst)            = pack_bf16(r0.x, r1.x);    // [w2  ][c,c+1]
        *reinterpret_cast<unsigned*>(dst + LDS_ROW)  = pack_bf16(r0.y, r1.y);    // [w2+1][c,c+1]
    }
    __syncthreads();

    // exp2-domain scale: (1/sqrt(128)) * log2(e), so v_exp_f32 gets its input
    // with a single multiply; masked max is tracked in the same domain.
    const float k2 = 0.12751879526536455f;

    // ---- per-wave: two 16-row M-stripes at a time; each B fragment feeds 2 WMMAs ----
    for (int pass = 0; pass < 2; ++pass) {
        const int stripe0 = (wave +      16 * pass) * 16;   // rows stripe0..stripe0+15
        const int stripe1 = (wave + 8  + 16 * pass) * 16;   // rows stripe1..stripe1+15
        const int mrow0   = stripe0 + (lane & 15);
        const int mrow1   = stripe1 + (lane & 15);

        // A fragments (16x32 bf16 each) for all 4 K-chunks, both stripes.
        // Layout (ISA 7.12.2): half 0: K = kc*32+[0..7],[16..23]; half 1: +8.
        bf16x16 afrag0[4], afrag1[4];
        #pragma unroll
        for (int kc = 0; kc < 4; ++kc) {
            const char* p0 = smem + A_BASE + (size_t)mrow0 * LDS_ROW + kc * 64 + half * 16;
            const char* p1 = smem + A_BASE + (size_t)mrow1 * LDS_ROW + kc * 64 + half * 16;
            V16 u0, u1;
            u0.q[0] = *reinterpret_cast<const uint4*>(p0);
            u0.q[1] = *reinterpret_cast<const uint4*>(p0 + 32);
            u1.q[0] = *reinterpret_cast<const uint4*>(p1);
            u1.q[1] = *reinterpret_cast<const uint4*>(p1 + 32);
            afrag0[kc] = u0.v;
            afrag1[kc] = u1.v;
        }

        // running stats per accumulator VGPR g (row i = stripe + g + 8*half)
        float ssum0[8], twt0[8], vmax0[8];
        float ssum1[8], twt1[8], vmax1[8];
        #pragma unroll
        for (int g = 0; g < 8; ++g) {
            ssum0[g] = 0.f; twt0[g] = 0.f; vmax0[g] = -INFINITY;
            ssum1[g] = 0.f; twt1[g] = 0.f; vmax1[g] = -INFINITY;
        }
        const int irow0 = stripe0 + (half ? 8 : 0);
        const int irow1 = stripe1 + (half ? 8 : 0);

        for (int nt = 0; nt < 32; ++nt) {
            const int ncol = nt * 16 + (lane & 15);   // column j this lane holds in C/D
            f32x8 acc0 = {0.f, 0.f, 0.f, 0.f, 0.f, 0.f, 0.f, 0.f};
            f32x8 acc1 = {0.f, 0.f, 0.f, 0.f, 0.f, 0.f, 0.f, 0.f};
            #pragma unroll
            for (int kc = 0; kc < 4; ++kc) {
                // B fragment (32x16 bf16): half 0: K = kc*32+[0..7],[16..23]; half 1: +8
                const char* p = smem + B_BASE + (size_t)ncol * LDS_ROW + kc * 64 + half * 16;
                V16 u;
                u.q[0] = *reinterpret_cast<const uint4*>(p);
                u.q[1] = *reinterpret_cast<const uint4*>(p + 32);
                acc0 = __builtin_amdgcn_wmma_f32_16x16x32_bf16(
                           false, afrag0[kc], false, u.v, (short)0, acc0, false, false);
                acc1 = __builtin_amdgcn_wmma_f32_16x16x32_bf16(
                           false, afrag1[kc], false, u.v, (short)0, acc1, false, false);
            }
            const float jf = (float)ncol;
            #pragma unroll
            for (int g = 0; g < 8; ++g) {
                float v0 = acc0[g] * k2;                 // log2-domain score
                float v1 = acc1[g] * k2;
                float e0 = fast_exp2(v0);
                float e1 = fast_exp2(v1);
                ssum0[g] += e0;                          // softmax denominator (all j)
                ssum1[g] += e1;
                bool m0 = (ncol <= irow0 + g);           // tril mask j <= i
                bool m1 = (ncol <= irow1 + g);
                twt0[g] += (m0 ? e0 : 0.f) * jf;         // sum p*j (masked)
                twt1[g] += (m1 ? e1 : 0.f) * jf;
                vmax0[g] = m0 ? fmaxf(vmax0[g], v0) : vmax0[g];  // masked max (log2 dom.)
                vmax1[g] = m1 ? fmaxf(vmax1[g], v1) : vmax1[g];
            }
        }

        // reduce across the 16 lanes that share each row, then finalize + store
        #pragma unroll
        for (int st = 0; st < 2; ++st) {
            const int   irow = st ? irow1 : irow0;
            float*      ssum = st ? ssum1 : ssum0;
            float*      twt  = st ? twt1  : twt0;
            float*      vmax = st ? vmax1 : vmax0;
            #pragma unroll
            for (int g = 0; g < 8; ++g) {
                float ss = ssum[g], tt = twt[g], mm = vmax[g];
                #pragma unroll
                for (int off = 8; off >= 1; off >>= 1) {
                    ss += __shfl_xor(ss, off);
                    tt += __shfl_xor(tt, off);
                    mm  = fmaxf(mm, __shfl_xor(mm, off));
                }
                if ((lane & 15) == g) {
                    const int i = irow + g;
                    float corresp = tt / ss;
                    float conf    = fast_exp2(mm) / ss;     // max prob on the row
                    float disp    = fabsf(corresp - (float)i) * (1.0f / (float)W_);
                    disp = fmaxf(disp, 0.1f);
                    out[((size_t)(b * 2 + 0) * H_ + h) * W_ + i] = fxb_over_far / disp;
                    out[((size_t)(b * 2 + 1) * H_ + h) * W_ + i] = conf;
                }
            }
        }
    }
}

extern "C" void kernel_launch(void* const* d_in, const int* in_sizes, int n_in,
                              void* d_out, int out_size, void* d_ws, size_t ws_size,
                              hipStream_t stream) {
    const float* feat  = (const float*)d_in[0];   // (4,2,128,128,512) f32
    const float* extri = (const float*)d_in[1];   // (4,2,4,4)
    const float* intri = (const float*)d_in[2];   // (4,2,3,3)
    // d_in[3] = near (unused by reference)
    const float* farp  = (const float*)d_in[4];   // (4,2)
    float* out = (float*)d_out;                   // (4,1,2,128,512)

    (void)in_sizes; (void)n_in; (void)out_size; (void)d_ws; (void)ws_size;

    (void)hipFuncSetAttribute((const void*)costvol_fused_kernel,
                              hipFuncAttributeMaxDynamicSharedMemorySize, SMEM_BYTES);

    costvol_fused_kernel<<<dim3(512), dim3(256), SMEM_BYTES, stream>>>(
        feat, extri, intri, farp, out);
}
